// TSP_GGCN_v2_48395691491780
// MI455X (gfx1250) — matrix operand order repaired and checked by reading, not compiled, not run on hardware
//
#include <hip/hip_runtime.h>
#include <hip/hip_bf16.h>

typedef __attribute__((ext_vector_type(16))) _Float16 v16h;
typedef __attribute__((ext_vector_type(8)))  float    v8f;

// ---------------------------------------------------------------------------
// WMMA fragment helpers (CDNA5 v_wmma_f32_16x16x32_f16, wave32)
// A 16x32 f16 (ISA 7.12.2): lanes 0-15 -> M=lane, K in {koff..koff+7,
//   16+koff..16+koff+7}, koff = 0 (lanes<16) / 8 (lanes>=16).
// B 32x16 f16: lane gives N = lane&15; K same order as A:
//   element e of lane l -> k = (e<8 ? koff+e : 16+koff+e-8), koff = (l<16?0:8).
// C/D 16x16 f32: elem j -> row = j + (lane<16?0:8), col = lane&15.
// B matrices are PRE-PACKED into fragment order:
//   packed[((kt*NT + nt)*32 + lane)*16 + e]  ->  two contiguous b128 per lane.
// ---------------------------------------------------------------------------
__device__ __forceinline__ v16h load_afrag(const _Float16* base, int ld, int row0, int k0) {
  const int lane = threadIdx.x & 31;
  const int m = lane & 15;
  const int koff = (lane < 16) ? 0 : 8;
  const _Float16* p = base + (long long)(row0 + m) * ld + k0 + koff;
  v16h a;
#pragma unroll
  for (int i = 0; i < 8; ++i) a[i] = p[i];
#pragma unroll
  for (int i = 0; i < 8; ++i) a[8 + i] = p[16 + i];
  return a;
}

__device__ __forceinline__ v16h load_bfrag_packed(const _Float16* base, int nTiles,
                                                  int kt, int nt) {
  const int lane = threadIdx.x & 31;
  const _Float16* p = base + ((long long)((kt * nTiles + nt) * 32 + lane) << 4);
  v16h b;
#pragma unroll
  for (int i = 0; i < 16; ++i) b[i] = p[i];
  return b;
}

__device__ __forceinline__ v8f wmma_f16(v16h a, v16h b, v8f c) {
  return __builtin_amdgcn_wmma_f32_16x16x32_f16(false, a, false, b, (short)0, c, false, false);
}

__device__ __forceinline__ float sigmf(float x) { return 1.0f / (1.0f + __expf(-x)); }

// packed B index -> logical (k, n)
__device__ __forceinline__ void frag_kn(int i, int nTiles, int& k, int& n) {
  int tile = i >> 9;            // / 512
  int within = i & 511;
  int lane = within >> 4, e = within & 15;
  int kt = tile / nTiles, nt = tile - kt * nTiles;
  k = kt * 32 + ((e & 8) << 1) + ((lane & 16) ? 8 : 0) + (e & 7);
  n = nt * 16 + (lane & 15);
}

// ---------------------------------------------------------------------------
// Elementwise / packing kernels
// ---------------------------------------------------------------------------
__global__ void k_zero_f32(float* p, long long n) {
  long long i = blockIdx.x * (long long)blockDim.x + threadIdx.x;
  if (i < n) p[i] = 0.0f;
}

__global__ void k_cast_f16(const float* __restrict__ s, _Float16* __restrict__ d, long long n) {
  long long i = blockIdx.x * (long long)blockDim.x + threadIdx.x;
  if (i < n) d[i] = (_Float16)s[i];
}

// features [N,2] -> h [N,64] fp32 + f16 (zero padded)
__global__ void k_init_h(const float* __restrict__ feat, float* __restrict__ h,
                         _Float16* __restrict__ h16, int N) {
  long long i = blockIdx.x * (long long)blockDim.x + threadIdx.x;
  if (i >= (long long)N * 64) return;
  int n = (int)(i >> 6), c = (int)(i & 63);
  float v = (c < 2) ? feat[n * 2 + c] : 0.0f;
  h[i] = v; h16[i] = (_Float16)v;
}

// relu + repad: src [N,CPold] (valid cols < Cvalid) -> dst [N,CPnew] fp32+f16
__global__ void k_relu_pad(const float* __restrict__ src, int Cvalid, int CPold,
                           float* __restrict__ dst, _Float16* __restrict__ dst16,
                           int CPnew, int N) {
  long long i = blockIdx.x * (long long)blockDim.x + threadIdx.x;
  if (i >= (long long)N * CPnew) return;
  int n = (int)(i / CPnew), c = (int)(i % CPnew);
  float v = 0.0f;
  if (c < Cvalid) { v = fmaxf(src[(long long)n * CPold + c], 0.0f); }
  dst[i] = v; dst16[i] = (_Float16)v;
}

// conv weight[l] (C x C, row=k, col=n) -> packed frag f16 [CP x CP]
__global__ void k_pack_w(const float* __restrict__ src, _Float16* __restrict__ dst, int C, int CP) {
  int i = blockIdx.x * blockDim.x + threadIdx.x;
  if (i >= CP * CP) return;
  int k, n; frag_kn(i, CP >> 4, k, n);
  dst[i] = (k < C && n < C) ? (_Float16)src[k * C + n] : (_Float16)0.0f;
}

// GRU weight (3C x C) -> packed frag f16 [CP x 3CP], per-gate padded, transposed:
// eff[k][g*CP+j] = src[(g*C+j)*C + k]
__global__ void k_pack_gruw(const float* __restrict__ src, _Float16* __restrict__ dst, int C, int CP) {
  int i = blockIdx.x * blockDim.x + threadIdx.x;
  if (i >= CP * 3 * CP) return;
  int k, n; frag_kn(i, (3 * CP) >> 4, k, n);
  int g = n / CP, j = n - g * CP;
  float v = (k < C && j < C) ? src[(g * C + j) * C + k] : 0.0f;
  dst[i] = (_Float16)v;
}

// FC weight (Nsrc x Ksrc) -> packed frag f16 [KP x NP], transposed
__global__ void k_pack_fcT(const float* __restrict__ src, _Float16* __restrict__ dst,
                           int Ksrc, int Nsrc, int KP, int NP) {
  int i = blockIdx.x * blockDim.x + threadIdx.x;
  if (i >= KP * NP) return;
  int k, n; frag_kn(i, NP >> 4, k, n);
  float v = (k < Ksrc && n < Nsrc) ? src[n * Ksrc + k] : 0.0f;
  dst[i] = (_Float16)v;
}

// fc_edges_1 weight (100 x 300) -> packed frag f16 [320 x 128]; concat layout:
// k in [0,160): src half (kk=k), k in [160,320): dst half (kk=k-160)
__global__ void k_pack_fc1(const float* __restrict__ src, _Float16* __restrict__ dst) {
  int i = blockIdx.x * blockDim.x + threadIdx.x;
  if (i >= 320 * 128) return;
  int k, n; frag_kn(i, 8, k, n);
  int part = (k < 160) ? 0 : 1;
  int kk = k - part * 160;
  float v = (kk < 150 && n < 100) ? src[n * 300 + part * 150 + kk] : 0.0f;
  dst[i] = (_Float16)v;
}

// bias pad: src [G*C] gate-major -> dst [G*CP]
__global__ void k_pack_bias(const float* __restrict__ src, float* __restrict__ dst,
                            int C, int CP, int G) {
  int i = blockIdx.x * blockDim.x + threadIdx.x;
  if (i >= G * CP) return;
  int g = i / CP, j = i % CP;
  dst[i] = (j < C) ? src[g * C + j] : 0.0f;
}

// ---------------------------------------------------------------------------
// Row-tiled WMMA GEMM (templated, fully unrolled, register-blocked):
// OUT[M,NP] = A[M,K]f16 @ Bpacked[K,NP]f16 (+bias)(relu)
// block = 128 (4 waves), 64 rows/block, A staged in LDS.
// M % 16 == 0 -> wave tiles never straddle M (uniform guard).
// ---------------------------------------------------------------------------
template <int K, int NP, bool BIAS, bool RELU, bool OUTH>
__global__ void __launch_bounds__(128) k_gemm_rows(
    const _Float16* __restrict__ A, const _Float16* __restrict__ Bp,
    const float* __restrict__ bias, float* __restrict__ outF,
    _Float16* __restrict__ outH, int M) {
  constexpr int NT = NP / 16, KT = K / 32, H2 = K / 2;
  __shared__ _Float16 sA[64 * K];
  const int row0 = blockIdx.x * 64;
  for (int i = threadIdx.x; i < 64 * H2; i += 128) {
    int r = i / H2, c = i - r * H2;
    int gr = row0 + r;
    unsigned v = 0u;
    if (gr < M) v = ((const unsigned*)(A + (long long)gr * K))[c];
    ((unsigned*)sA)[i] = v;
  }
  __syncthreads();
  const int w = threadIdx.x >> 5, lane = threadIdx.x & 31;
  const int r0 = w * 16;
  if (row0 + r0 >= M) return;           // wave-uniform tile guard
  v8f acc[NT];
  const v8f vzero = {};
#pragma unroll
  for (int nt = 0; nt < NT; ++nt) acc[nt] = vzero;
#pragma unroll
  for (int kt = 0; kt < KT; ++kt) {
    v16h a = load_afrag(sA, K, r0, kt * 32);
#pragma unroll
    for (int nt = 0; nt < NT; ++nt) {
      v16h b = load_bfrag_packed(Bp, NT, kt, nt);
      acc[nt] = wmma_f16(a, b, acc[nt]);
    }
  }
  const int rbase = row0 + r0 + ((lane < 16) ? 0 : 8);
#pragma unroll
  for (int nt = 0; nt < NT; ++nt) {
    const int col = nt * 16 + (lane & 15);
    const float bv = BIAS ? bias[col] : 0.0f;
#pragma unroll
    for (int j = 0; j < 8; ++j) {
      float v = acc[nt][j] + bv;
      if (RELU) v = fmaxf(v, 0.0f);
      outF[(long long)(rbase + j) * NP + col] = v;
      if (OUTH) outH[(long long)(rbase + j) * NP + col] = (_Float16)v;
    }
  }
}

// ---------------------------------------------------------------------------
// Edge message: agg[dst] += m[src] * ew  (L2-resident fp32 atomics)
// grid.x covers edges, grid.y covers 4-channel chunks.
// ---------------------------------------------------------------------------
__global__ void k_edge_msg(const float* __restrict__ m, const long long* __restrict__ src,
                           const long long* __restrict__ dst, const float* __restrict__ ew,
                           float* __restrict__ agg, int CP, int E) {
  int e = blockIdx.x * blockDim.x + threadIdx.x;
  if (e >= E) return;
  int c4 = blockIdx.y << 2;
  long long s = src[e], d = dst[e];
  float w = ew[e];
  const float4 v = *(const float4*)(m + s * CP + c4);
  float* ap = agg + d * CP + c4;
  atomicAdd(ap + 0, v.x * w);
  atomicAdd(ap + 1, v.y * w);
  atomicAdd(ap + 2, v.z * w);
  atomicAdd(ap + 3, v.w * w);
}

// ---------------------------------------------------------------------------
// Fused GRU (templated on CP): gi = agg @ WihT + bih ; gh = h @ WhhT + bhh ;
// gate math ; h update.  block = 128 (4 waves), 64 rows/block.
// ---------------------------------------------------------------------------
template <int CP>
__global__ void __launch_bounds__(128) k_gru(
    const _Float16* __restrict__ agg16, const _Float16* __restrict__ h16,
    float* __restrict__ h, _Float16* __restrict__ h16out,
    const _Float16* __restrict__ Wih, const _Float16* __restrict__ Whh,
    const float* __restrict__ bih, const float* __restrict__ bhh, int M) {
  constexpr int JT = CP / 16, KT = CP / 32, NT3 = 3 * CP / 16, H2 = CP / 2;
  __shared__ _Float16 sAgg[64 * CP];
  __shared__ _Float16 sH[64 * CP];
  const int row0 = blockIdx.x * 64;
  for (int i = threadIdx.x; i < 64 * H2; i += 128) {
    int r = i / H2, c = i - r * H2;
    int gr = row0 + r;
    unsigned va = 0u, vh = 0u;
    if (gr < M) {
      va = ((const unsigned*)(agg16 + (long long)gr * CP))[c];
      vh = ((const unsigned*)(h16 + (long long)gr * CP))[c];
    }
    ((unsigned*)sAgg)[i] = va;
    ((unsigned*)sH)[i] = vh;
  }
  __syncthreads();
  const int w = threadIdx.x >> 5, lane = threadIdx.x & 31;
  const int r0 = w * 16;
  if (row0 + r0 >= M) return;           // wave-uniform tile guard
  for (int jt = 0; jt < JT; ++jt) {
    v8f ai0 = {}, ai1 = {}, ai2 = {}, ah0 = {}, ah1 = {}, ah2 = {};
#pragma unroll
    for (int kt = 0; kt < KT; ++kt) {
      v16h aA = load_afrag(sAgg, CP, r0, kt * 32);
      v16h aH = load_afrag(sH, CP, r0, kt * 32);
      v16h b;
      b = load_bfrag_packed(Wih, NT3, kt, 0 * JT + jt); ai0 = wmma_f16(aA, b, ai0);
      b = load_bfrag_packed(Wih, NT3, kt, 1 * JT + jt); ai1 = wmma_f16(aA, b, ai1);
      b = load_bfrag_packed(Wih, NT3, kt, 2 * JT + jt); ai2 = wmma_f16(aA, b, ai2);
      b = load_bfrag_packed(Whh, NT3, kt, 0 * JT + jt); ah0 = wmma_f16(aH, b, ah0);
      b = load_bfrag_packed(Whh, NT3, kt, 1 * JT + jt); ah1 = wmma_f16(aH, b, ah1);
      b = load_bfrag_packed(Whh, NT3, kt, 2 * JT + jt); ah2 = wmma_f16(aH, b, ah2);
    }
    const int col = jt * 16 + (lane & 15);
    const int rbase = row0 + r0 + ((lane < 16) ? 0 : 8);
    const float br = bih[col], bz = bih[CP + col], bn = bih[2 * CP + col];
    const float cr = bhh[col], cz = bhh[CP + col], cn = bhh[2 * CP + col];
#pragma unroll
    for (int j = 0; j < 8; ++j) {
      float r = sigmf(ai0[j] + br + ah0[j] + cr);
      float z = sigmf(ai1[j] + bz + ah1[j] + cz);
      float nt = tanhf(ai2[j] + bn + r * (ah2[j] + cn));
      long long off = (long long)(rbase + j) * CP + col;
      float hold = h[off];
      float hv = (1.0f - z) * nt + z * hold;
      h[off] = hv;
      h16out[off] = (_Float16)hv;
    }
  }
}

// ---------------------------------------------------------------------------
// Fused edge MLP: gather concat(h[src],h[dst]) -> GEMM1(320->128) -> GEMM2(128->2)
// block = 128 (4 waves), 16 edges per wave.  E % 64 == 0.
// ---------------------------------------------------------------------------
__global__ void __launch_bounds__(128) k_edge_mlp(
    const _Float16* __restrict__ hf16, const long long* __restrict__ src,
    const long long* __restrict__ dst, const _Float16* __restrict__ W1,
    const float* __restrict__ b1, const _Float16* __restrict__ W2,
    const float* __restrict__ b2, float* __restrict__ out, int E) {
  __shared__ _Float16 sCat[4 * 16 * 320];
  __shared__ _Float16 sHid[4 * 16 * 128];
  const int w = threadIdx.x >> 5, lane = threadIdx.x & 31;
  const int e0 = (blockIdx.x * 4 + w) * 16;
  // gather: 2 lanes per edge; each copies one 160-half row (320B = 80 dwords)
  {
    int i = lane >> 1, part = lane & 1;
    long long eidx = e0 + i;
    long long node = part ? dst[eidx] : src[eidx];
    const unsigned* sp = (const unsigned*)(hf16 + node * 160);
    unsigned* dp = (unsigned*)(sCat + w * 16 * 320 + i * 320 + part * 160);
#pragma unroll 8
    for (int c = 0; c < 80; ++c) dp[c] = sp[c];
  }
  __syncthreads();
  _Float16* myCat = sCat + w * 16 * 320;
  _Float16* myHid = sHid + w * 16 * 128;
  // GEMM1: [16,320] @ [320,128] + b1 -> f16 LDS  (register-blocked, 8 acc)
  {
    v8f acc[8];
    const v8f vzero = {};
#pragma unroll
    for (int nt = 0; nt < 8; ++nt) acc[nt] = vzero;
#pragma unroll
    for (int kt = 0; kt < 10; ++kt) {
      v16h a = load_afrag(myCat, 320, 0, kt * 32);
#pragma unroll
      for (int nt = 0; nt < 8; ++nt) {
        v16h b = load_bfrag_packed(W1, 8, kt, nt);
        acc[nt] = wmma_f16(a, b, acc[nt]);
      }
    }
    const int rb = (lane < 16) ? 0 : 8;
#pragma unroll
    for (int nt = 0; nt < 8; ++nt) {
      int col = nt * 16 + (lane & 15);
      float bv = b1[col];
#pragma unroll
      for (int j = 0; j < 8; ++j)
        myHid[(rb + j) * 128 + col] = (_Float16)(acc[nt][j] + bv);
    }
  }
  __syncthreads();
  // GEMM2: [16,128] @ [128,16] + b2 -> out cols 0,1
  v8f acc = {};
#pragma unroll
  for (int kt = 0; kt < 4; ++kt) {
    v16h a = load_afrag(myHid, 128, 0, kt * 32);
    v16h b = load_bfrag_packed(W2, 1, kt, 0);
    acc = wmma_f16(a, b, acc);
  }
  const int col = lane & 15;
  const int rb = (lane < 16) ? 0 : 8;
  if (col < 2) {
    float bv = b2[col];
#pragma unroll
    for (int j = 0; j < 8; ++j) {
      long long e = e0 + rb + j;
      out[e * 2 + col] = acc[j] + bv;
    }
  }
}

// ---------------------------------------------------------------------------
// Host orchestration
// ---------------------------------------------------------------------------
extern "C" void kernel_launch(void* const* d_in, const int* in_sizes, int n_in,
                              void* d_out, int out_size, void* d_ws, size_t ws_size,
                              hipStream_t stream) {
  (void)in_sizes; (void)n_in; (void)out_size; (void)ws_size;
  const int N = 50000, E = 800000;
  const int Cs[3] = {50, 100, 150}, CPs[3] = {64, 128, 160};

  const float* features = (const float*)d_in[0];
  const long long* ei = (const long long*)d_in[1];
  const float* dist = (const float*)d_in[2];
  const long long* esrc = ei;
  const long long* edst = ei + E;
  // params flattened in pytree (sorted-key) order
  const float *bhh[3], *bih[3], *whh[3], *wih[3], *wgt[3];
  for (int i = 0; i < 3; ++i) {
    int b = 3 + i * 5;
    bhh[i] = (const float*)d_in[b];     bih[i] = (const float*)d_in[b + 1];
    whh[i] = (const float*)d_in[b + 2]; wih[i] = (const float*)d_in[b + 3];
    wgt[i] = (const float*)d_in[b + 4];
  }
  const float* fce1_b = (const float*)d_in[18];
  const float* fce1_w = (const float*)d_in[19];
  const float* fce2_b = (const float*)d_in[20];
  const float* fce2_w = (const float*)d_in[21];
  const float* fcn_b  = (const float*)d_in[22];
  const float* fcn_w  = (const float*)d_in[23];

  // workspace carving
  char* wsp = (char*)d_ws;
  size_t off = 0;
  auto carve = [&](size_t bytes) -> char* {
    char* p = wsp + off; off += (bytes + 255) & ~(size_t)255; return p;
  };
  float* hA = (float*)carve((size_t)N * 160 * 4);
  float* hB = (float*)carve((size_t)N * 160 * 4);
  float* mbuf = (float*)carve((size_t)N * 160 * 4);
  float* agg = (float*)carve((size_t)N * 160 * 4);
  _Float16* hA16 = (_Float16*)carve((size_t)N * 160 * 2);
  _Float16* hB16 = (_Float16*)carve((size_t)N * 160 * 2);
  _Float16* agg16 = (_Float16*)carve((size_t)N * 160 * 2);
  _Float16* wl16[3][3]; _Float16* wih16[3]; _Float16* whh16[3];
  float* bihp[3]; float* bhhp[3];
  for (int i = 0; i < 3; ++i) {
    int CP = CPs[i];
    for (int l = 0; l < 3; ++l) wl16[i][l] = (_Float16*)carve((size_t)CP * CP * 2);
    wih16[i] = (_Float16*)carve((size_t)CP * 3 * CP * 2);
    whh16[i] = (_Float16*)carve((size_t)CP * 3 * CP * 2);
    bihp[i] = (float*)carve((size_t)3 * CP * 4);
    bhhp[i] = (float*)carve((size_t)3 * CP * 4);
  }
  _Float16* fcnW16 = (_Float16*)carve(160 * 160 * 2);
  _Float16* fc1W16 = (_Float16*)carve(320 * 128 * 2);
  _Float16* fc2W16 = (_Float16*)carve(128 * 16 * 2);
  float* fcnBp = (float*)carve(160 * 4);
  float* fc1Bp = (float*)carve(128 * 4);
  float* fc2Bp = (float*)carve(16 * 4);

  auto g1 = [](long long n) { return dim3((unsigned)((n + 255) / 256)); };

  // pack all params into fragment-swizzled f16 / padded fp32 bias
  for (int i = 0; i < 3; ++i) {
    int C = Cs[i], CP = CPs[i];
    for (int l = 0; l < 3; ++l)
      k_pack_w<<<g1((long long)CP * CP), 256, 0, stream>>>(wgt[i] + (size_t)l * C * C, wl16[i][l], C, CP);
    k_pack_gruw<<<g1((long long)CP * 3 * CP), 256, 0, stream>>>(wih[i], wih16[i], C, CP);
    k_pack_gruw<<<g1((long long)CP * 3 * CP), 256, 0, stream>>>(whh[i], whh16[i], C, CP);
    k_pack_bias<<<g1(3 * CP), 256, 0, stream>>>(bih[i], bihp[i], C, CP, 3);
    k_pack_bias<<<g1(3 * CP), 256, 0, stream>>>(bhh[i], bhhp[i], C, CP, 3);
  }
  k_pack_fcT<<<g1(160 * 160), 256, 0, stream>>>(fcn_w, fcnW16, 150, 150, 160, 160);
  k_pack_fc1<<<g1(320 * 128), 256, 0, stream>>>(fce1_w, fc1W16);
  k_pack_fcT<<<g1(128 * 16), 256, 0, stream>>>(fce2_w, fc2W16, 100, 2, 128, 16);
  k_pack_bias<<<g1(160), 256, 0, stream>>>(fcn_b, fcnBp, 150, 160, 1);
  k_pack_bias<<<g1(128), 256, 0, stream>>>(fce1_b, fc1Bp, 100, 128, 1);
  k_pack_bias<<<g1(16), 256, 0, stream>>>(fce2_b, fc2Bp, 2, 16, 1);

  k_init_h<<<g1((long long)N * 64), 256, 0, stream>>>(features, hA, hA16, N);

  float* hF = hA; _Float16* hH = hA16;
  float* hFo = hB; _Float16* hHo = hB16;
  const dim3 blk128(128);
  const int rowBlocks = (N + 63) / 64;

  auto gemm_plain = [&](const _Float16* A, const _Float16* Bp, float* outF, int CP) {
    if (CP == 64)
      k_gemm_rows<64, 64, false, false, false><<<rowBlocks, blk128, 0, stream>>>(
          A, Bp, nullptr, outF, nullptr, N);
    else if (CP == 128)
      k_gemm_rows<128, 128, false, false, false><<<rowBlocks, blk128, 0, stream>>>(
          A, Bp, nullptr, outF, nullptr, N);
    else
      k_gemm_rows<160, 160, false, false, false><<<rowBlocks, blk128, 0, stream>>>(
          A, Bp, nullptr, outF, nullptr, N);
  };
  auto gru_step = [&](int i, int CP) {
    if (CP == 64)
      k_gru<64><<<rowBlocks, blk128, 0, stream>>>(agg16, hH, hF, hH, wih16[i], whh16[i],
                                                  bihp[i], bhhp[i], N);
    else if (CP == 128)
      k_gru<128><<<rowBlocks, blk128, 0, stream>>>(agg16, hH, hF, hH, wih16[i], whh16[i],
                                                   bihp[i], bhhp[i], N);
    else
      k_gru<160><<<rowBlocks, blk128, 0, stream>>>(agg16, hH, hF, hH, wih16[i], whh16[i],
                                                   bihp[i], bhhp[i], N);
  };

  for (int i = 0; i < 3; ++i) {
    int C = Cs[i], CP = CPs[i];
    for (int l = 0; l < 3; ++l) {
      // m = h @ W[l]
      gemm_plain(hH, wl16[i][l], mbuf, CP);
      // agg = segment_sum(m[src] * ew, dst)
      k_zero_f32<<<g1((long long)N * CP), 256, 0, stream>>>(agg, (long long)N * CP);
      {
        dim3 grid((unsigned)((E + 255) / 256), (unsigned)(CP / 4));
        k_edge_msg<<<grid, 256, 0, stream>>>(mbuf, esrc, edst, dist, agg, CP, E);
      }
      k_cast_f16<<<g1((long long)N * CP), 256, 0, stream>>>(agg, agg16, (long long)N * CP);
      // h = GRUCell(agg, h)
      gru_step(i, CP);
    }
    int CPnew = (i < 2) ? CPs[i + 1] : 160;
    k_relu_pad<<<g1((long long)N * CPnew), 256, 0, stream>>>(hF, C, CP, hFo, hHo, CPnew, N);
    float* tF = hF; hF = hFo; hFo = tF;
    _Float16* tH = hH; hH = hHo; hHo = tH;
  }

  // hfinal = relu(h @ Wfc^T + b)   (fp32 + f16 copy for edge gather)
  k_gemm_rows<160, 160, true, true, true><<<rowBlocks, blk128, 0, stream>>>(
      hH, fcnW16, fcnBp, hFo, hHo, N);

  // edge MLP -> d_out [E,2]
  k_edge_mlp<<<dim3(E / 64), blk128, 0, stream>>>(hHo, esrc, edst, fc1W16, fc1Bp,
                                                  fc2W16, fc2Bp, (float*)d_out, E);
}
